// VariationalLinear_5574867550300
// MI455X (gfx1250) — matrix operand order repaired and checked by reading, not compiled, not run on hardware
//
#include <hip/hip_runtime.h>
#include <stdint.h>

#define IN_DIM  4096
#define OUT_DIM 4096
#define BATCH_N 512

typedef __attribute__((ext_vector_type(16))) __bf16 v16bf;
typedef __attribute__((ext_vector_type(8)))  float  v8f;

__device__ __forceinline__ uint16_t f32_to_bf16(float f) {
  uint32_t u = __float_as_uint(f);
  uint32_t r = u + 0x7FFFu + ((u >> 16) & 1u);   // round-to-nearest-even
  return (uint16_t)(r >> 16);
}

// ---------------------------------------------------------------------------
// Kernel 1: sp = softplus(sigma); w = mu + sp*eps; store w as bf16;
// per-element KL contribution q - p, block-reduced into partials[].
// Each block handles 4096 contiguous elements (256 threads x 16, coalesced).
// ---------------------------------------------------------------------------
__global__ void vl_weight_kl_kernel(const float* __restrict__ wmu,
                                    const float* __restrict__ wsig,
                                    const float* __restrict__ eps,
                                    uint16_t* __restrict__ wbf,
                                    float* __restrict__ partials) {
  const int tid = threadIdx.x;
  const size_t base = (size_t)blockIdx.x * 4096;
  float local = 0.f;
  #pragma unroll 4
  for (int e = 0; e < 16; ++e) {
    size_t i = base + (size_t)e * 256 + tid;
    float ws = wsig[i];
    float sp = (ws > 20.f) ? ws : log1pf(__expf(ws));   // softplus
    float ep = eps[i];
    float w  = wmu[i] + sp * ep;
    wbf[i] = f32_to_bf16(w);
    // q_log_prob: -((mu-w)^2)/(2 sp) - 0.5 log sp ; (mu-w) = -sp*ep
    float q = -0.5f * sp * ep * ep - 0.5f * __logf(sp);
    // mixture prior: both components share exp(-w^2)
    // g(w,s) = 1/(s*2.506) * exp(-w^2) / (2 s^2)
    const float C1 = 0.5f * (1.0f / (1.0f * 2.506f)) / (2.0f * 1.0f * 1.0f);
    const float C2 = 0.5f * (1.0f / (0.1f * 2.506f)) / (2.0f * 0.1f * 0.1f);
    float p = __logf(C1 + C2) - w * w;
    local += q - p;
  }
  __shared__ float red[256];
  red[tid] = local;
  __syncthreads();
  for (int s = 128; s > 0; s >>= 1) {
    if (tid < s) red[tid] += red[tid + s];
    __syncthreads();
  }
  if (tid == 0) partials[blockIdx.x] = red[0];
}

// ---------------------------------------------------------------------------
// Kernel 2: x (f32) -> bf16
// ---------------------------------------------------------------------------
__global__ void vl_x_bf16_kernel(const float* __restrict__ x,
                                 uint16_t* __restrict__ xbf) {
  const size_t base = (size_t)blockIdx.x * 2048;
  #pragma unroll
  for (int e = 0; e < 8; ++e) {
    size_t i = base + (size_t)e * 256 + threadIdx.x;
    xbf[i] = f32_to_bf16(x[i]);
  }
}

// ---------------------------------------------------------------------------
// Kernel 3: reduce KL partials -> mean, write scalar at end of d_out
// ---------------------------------------------------------------------------
__global__ void vl_reduce_kernel(const float* __restrict__ partials, int n,
                                 float* __restrict__ out_kl) {
  __shared__ float red[256];
  float local = 0.f;
  for (int i = threadIdx.x; i < n; i += 256) local += partials[i];
  red[threadIdx.x] = local;
  __syncthreads();
  for (int s = 128; s > 0; s >>= 1) {
    if (threadIdx.x < s) red[threadIdx.x] += red[threadIdx.x + s];
    __syncthreads();
  }
  if (threadIdx.x == 0)
    *out_kl = red[0] * (1.0f / ((float)OUT_DIM * (float)IN_DIM));
}

// ---------------------------------------------------------------------------
// Kernel 4: WMMA bf16 GEMM  y[b,o] = sum_i x[b,i] * w[o,i]
// One wave computes a 16(M) x 64(N) tile: 4 f32 accumulators reusing A.
// Software-pipelined: fragments for step k+32 are loaded before the WMMAs
// of step k issue, so each WMMA consumes registers loaded a full iteration
// earlier (hides L2 latency instead of s_wait_loadcnt 0 before every WMMA).
//
// A fragment (16-bit A 16x32, contiguous K): lanes 0-15 -> M=lane, K{0..7} in
//   v0..3 and K{8..15} in v4..7; lanes 16-31 -> K{16..23},{24..31}.
// B fragment (32x16): lanes 0-15 -> N=lane, K=0..15 packed 2/VGPR in v0..7;
//   lanes 16-31 -> N=lane-16, K=16..31. Both map to row-major B128 loads.
// ---------------------------------------------------------------------------
__device__ __forceinline__ v16bf load_a_frag(const uint16_t* __restrict__ p) {
  union { v16bf v; uint4 q[2]; } u;
  u.q[0] = *(const uint4*)(p);        // K base+0..7   (or +16..23 upper half)
  u.q[1] = *(const uint4*)(p + 8);    // K base+8..15  (or +24..31)
  return u.v;
}

__device__ __forceinline__ v16bf load_b_frag(const uint16_t* __restrict__ p) {
  union { v16bf v; uint4 q[2]; } u;
  u.q[0] = *(const uint4*)(p);        // 32 contiguous bytes: K +0..15 / +16..31
  u.q[1] = *(const uint4*)(p + 8);
  return u.v;
}

__global__ void vl_gemm_kernel(const uint16_t* __restrict__ xb,
                               const uint16_t* __restrict__ wb,
                               float* __restrict__ y) {
  const int lane  = threadIdx.x & 31;
  const int wave  = (int)((blockIdx.x * blockDim.x + threadIdx.x) >> 5);
  const int nSup  = OUT_DIM / 64;           // 64 N-supertiles
  const int mTile = wave / nSup;            // 0..31
  const int nTile = wave % nSup;            // 0..63
  const int mbase = mTile * 16;
  const int obase = nTile * 64;
  const int half  = lane >> 4;              // 0 | 1
  const int r     = lane & 15;

  const uint16_t* xrow = xb + (size_t)(mbase + r) * IN_DIM + half * 16;
  const uint16_t* w0 = wb + (size_t)(obase +  0 + r) * IN_DIM + half * 16;
  const uint16_t* w1 = wb + (size_t)(obase + 16 + r) * IN_DIM + half * 16;
  const uint16_t* w2 = wb + (size_t)(obase + 32 + r) * IN_DIM + half * 16;
  const uint16_t* w3 = wb + (size_t)(obase + 48 + r) * IN_DIM + half * 16;

  v8f acc0 = {}, acc1 = {}, acc2 = {}, acc3 = {};

  // Prologue: load fragments for k = 0
  v16bf a  = load_a_frag(xrow);
  v16bf b0 = load_b_frag(w0);
  v16bf b1 = load_b_frag(w1);
  v16bf b2 = load_b_frag(w2);
  v16bf b3 = load_b_frag(w3);

  #pragma unroll 2
  for (int k = 0; k < IN_DIM - 32; k += 32) {
    // Issue next-step loads first so they overlap with this step's WMMAs.
    v16bf an  = load_a_frag(xrow + k + 32);
    v16bf b0n = load_b_frag(w0 + k + 32);
    v16bf b1n = load_b_frag(w1 + k + 32);
    v16bf b2n = load_b_frag(w2 + k + 32);
    v16bf b3n = load_b_frag(w3 + k + 32);

    acc0 = __builtin_amdgcn_wmma_f32_16x16x32_bf16(false, a, false, b0,
                                                   (short)0, acc0, false, false);
    acc1 = __builtin_amdgcn_wmma_f32_16x16x32_bf16(false, a, false, b1,
                                                   (short)0, acc1, false, false);
    acc2 = __builtin_amdgcn_wmma_f32_16x16x32_bf16(false, a, false, b2,
                                                   (short)0, acc2, false, false);
    acc3 = __builtin_amdgcn_wmma_f32_16x16x32_bf16(false, a, false, b3,
                                                   (short)0, acc3, false, false);
    a = an; b0 = b0n; b1 = b1n; b2 = b2n; b3 = b3n;
  }

  // Epilogue: final K step
  acc0 = __builtin_amdgcn_wmma_f32_16x16x32_bf16(false, a, false, b0,
                                                 (short)0, acc0, false, false);
  acc1 = __builtin_amdgcn_wmma_f32_16x16x32_bf16(false, a, false, b1,
                                                 (short)0, acc1, false, false);
  acc2 = __builtin_amdgcn_wmma_f32_16x16x32_bf16(false, a, false, b2,
                                                 (short)0, acc2, false, false);
  acc3 = __builtin_amdgcn_wmma_f32_16x16x32_bf16(false, a, false, b3,
                                                 (short)0, acc3, false, false);

  // C/D f32 16x16 layout: VGPR rr -> M = rr + 8*half ; N = obase + j*16 + r
  #pragma unroll
  for (int rr = 0; rr < 8; ++rr) {
    float* yrow = y + (size_t)(mbase + half * 8 + rr) * OUT_DIM + obase + r;
    yrow[0]  = acc0[rr];
    yrow[16] = acc1[rr];
    yrow[32] = acc2[rr];
    yrow[48] = acc3[rr];
  }
}

// ---------------------------------------------------------------------------
extern "C" void kernel_launch(void* const* d_in, const int* in_sizes, int n_in,
                              void* d_out, int out_size, void* d_ws, size_t ws_size,
                              hipStream_t stream) {
  const float* x    = (const float*)d_in[0];
  const float* wmu  = (const float*)d_in[1];
  const float* wsig = (const float*)d_in[2];
  const float* eps  = (const float*)d_in[3];

  float* y  = (float*)d_out;                       // [512, 4096]
  float* kl = y + (size_t)BATCH_N * OUT_DIM;       // scalar, appended

  char* ws = (char*)d_ws;
  uint16_t* wbf = (uint16_t*)ws;                                       // 32 MiB
  uint16_t* xbf = (uint16_t*)(ws + (size_t)OUT_DIM * IN_DIM * 2);      //  4 MiB
  float* partials = (float*)(ws + (size_t)OUT_DIM * IN_DIM * 2
                                + (size_t)BATCH_N * IN_DIM * 2);       // 16 KiB

  // 16.7M elements / (256 thr * 16 elems) = 4096 blocks
  vl_weight_kl_kernel<<<4096, 256, 0, stream>>>(wmu, wsig, eps, wbf, partials);
  // 2.1M elements / (256 * 8) = 1024 blocks
  vl_x_bf16_kernel<<<1024, 256, 0, stream>>>(x, xbf);
  vl_reduce_kernel<<<1, 256, 0, stream>>>(partials, 4096, kl);
  // 32 M-tiles * 64 N-supertiles = 2048 waves; 8 waves/block -> 256 blocks
  vl_gemm_kernel<<<256, 256, 0, stream>>>(xbf, wbf, y);
}
